// GenericMoeLayer_5952824672537
// MI455X (gfx1250) — matrix operand order repaired and checked by reading, not compiled, not run on hardware
//
#include <hip/hip_runtime.h>
#include <hip/hip_bf16.h>

// MoE top-4 of 16 experts, T=1024 H=1024 F=768, fp32 in/out.
// Strategy (MI455X / gfx1250, wave32):
//  - Sparse routing: only 4096 token-rows of GEMM total (19.3 GFLOP) instead of dense 77 GFLOP.
//  - Heavy GEMMs in v_wmma_f32_16x16x32_bf16 (f32 accumulate), converting f32->bf16 on load.
//    bf16 WMMA keeps the kernel memory-bound on the one-pass 150MB weight stream (6.9us @ 23.3TB/s)
//    instead of compute-bound on the slow 16x16x4 f32 WMMA path.
//  - Deterministic: expert lists built with ballot+prefix scan (token order), per-(token,slot)
//    row buffer instead of float atomics, fixed-order combine.

#define T_TOK 1024
#define H_DIM 1024
#define F_DIM 768
#define N_EXP 16
#define TOPK  4

typedef __attribute__((ext_vector_type(16))) __bf16 v16bf;
typedef __attribute__((ext_vector_type(8)))  float  v8f;
typedef __attribute__((ext_vector_type(8)))  unsigned v8u;

// ---------- helpers ----------

// f32 -> bf16 RNE pack of two values via v_perm_b32: round both into their
// high halves, then byte-permute {hi[31:16], lo[31:16]} in a single VALU op.
__device__ __forceinline__ unsigned bf16pk(float lo, float hi) {
  unsigned a = __builtin_bit_cast(unsigned, lo);
  unsigned b = __builtin_bit_cast(unsigned, hi);
  a += 0x7FFFu + ((a >> 16) & 1u);   // rounded bf16 now in a[31:16]
  b += 0x7FFFu + ((b >> 16) & 1u);
  return __builtin_amdgcn_perm(b, a, 0x07060302u);  // {b.b3,b.b2,a.b3,a.b2}
}

__device__ __forceinline__ v8f wmma_bf16(v16bf a, v16bf b, v8f c) {
  // 8 args: (neg_a, A, neg_b, B, c_mod, C, reuse_a, reuse_b)
  return __builtin_amdgcn_wmma_f32_16x16x32_bf16(false, a, false, b, (short)0, c,
                                                 false, false);
}

// A fragment (16x32 bf16, ISA 7.12.2): lane L -> row M = L%16, half h = L/16.
// VGPR i (pair of K): K = (i>=4?16:0) + 8h + 2*(i&3). LDS tile stored as
// uint pairs As[row][k/2], row stride 20 uints (80B, 16B aligned).
__device__ __forceinline__ v16bf lda_frag(const unsigned (*As)[20], int mbase, int lane) {
  int m = mbase + (lane & 15);
  int h = lane >> 4;
  uint4 x0 = *(const uint4*)&As[m][h * 4];
  uint4 x1 = *(const uint4*)&As[m][8 + h * 4];
  v8u u = {x0.x, x0.y, x0.z, x0.w, x1.x, x1.y, x1.z, x1.w};
  return __builtin_bit_cast(v16bf, u);
}

// B fragment (32x16 bf16): lane L -> col N = L%16, half h = L/16.
// VGPR j: K = (j>=4?16:0) + 8h + 2*(j&3). LDS tile stored K-transposed: Bt[n][k/2].
__device__ __forceinline__ v16bf ldb_frag(const unsigned (*Bt)[20], int nbase, int lane) {
  int n = nbase + (lane & 15);
  int h = lane >> 4;
  uint4 x0 = *(const uint4*)&Bt[n][h * 4];
  uint4 x1 = *(const uint4*)&Bt[n][8 + h * 4];
  v8u u = {x0.x, x0.y, x0.z, x0.w, x1.x, x1.y, x1.z, x1.w};
  return __builtin_bit_cast(v16bf, u);
}

__device__ __forceinline__ unsigned ballot32(bool p) {
#if __has_builtin(__builtin_amdgcn_ballot_w32)
  return __builtin_amdgcn_ballot_w32(p);
#else
  return (unsigned)__ballot(p);
#endif
}

// ---------- 1) router logits: logits[t,e] = dot(x[t,:], gate_w[e,:]) ----------

__global__ __launch_bounds__(256) void router_logits_k(
    const float* __restrict__ x, const float* __restrict__ gw, float* __restrict__ logits) {
  int idx = blockIdx.x * 256 + threadIdx.x;
  if (idx >= T_TOK * N_EXP) return;
  int t = idx >> 4, e = idx & 15;
  const float4* xp = (const float4*)(x + (size_t)t * H_DIM);
  const float4* gp = (const float4*)(gw + (size_t)e * H_DIM);
  float acc = 0.f;
#pragma unroll 4
  for (int i = 0; i < H_DIM / 4; ++i) {
    float4 a = xp[i], b = gp[i];
    acc += a.x * b.x + a.y * b.y + a.z * b.z + a.w * b.w;
  }
  logits[idx] = acc;
}

// ---------- 2) softmax + top-4 + renormalize ----------

__global__ __launch_bounds__(256) void router_topk_k(
    const float* __restrict__ logits, int* __restrict__ topk_id, float* __restrict__ topk_w) {
  int t = blockIdx.x * 256 + threadIdx.x;
  if (t >= T_TOK) return;
  float p[N_EXP];
  float mx = -3.4e38f;
#pragma unroll
  for (int e = 0; e < N_EXP; ++e) { p[e] = logits[t * N_EXP + e]; mx = fmaxf(mx, p[e]); }
  float sum = 0.f;
#pragma unroll
  for (int e = 0; e < N_EXP; ++e) { p[e] = __expf(p[e] - mx); sum += p[e]; }
  float inv = 1.f / sum;
#pragma unroll
  for (int e = 0; e < N_EXP; ++e) p[e] *= inv;
  int ids[TOPK]; float wts[TOPK]; float wsum = 0.f;
#pragma unroll
  for (int k = 0; k < TOPK; ++k) {
    int best = 0; float bv = -1.f;
#pragma unroll
    for (int e = 0; e < N_EXP; ++e)
      if (p[e] > bv) { bv = p[e]; best = e; }   // strict '>' => lowest index on ties (matches top_k)
    ids[k] = best; wts[k] = bv; wsum += bv; p[best] = -2.f;
  }
  float iw = 1.f / wsum;
#pragma unroll
  for (int k = 0; k < TOPK; ++k) {
    topk_id[t * TOPK + k] = ids[k];
    topk_w[t * TOPK + k] = wts[k] * iw;
  }
}

// ---------- 3) per-expert token lists (deterministic token-ordered scan) ----------

__global__ __launch_bounds__(1024) void build_lists_k(
    const int* __restrict__ topk_id, int* __restrict__ list, int* __restrict__ cnt) {
  int e = blockIdx.x;
  int t = threadIdx.x;            // one thread per token, 1024 = 32 waves
  int slot = -1;
#pragma unroll
  for (int s = 0; s < TOPK; ++s)
    if (topk_id[t * TOPK + s] == e) slot = s;
  bool flag = (slot >= 0);
  unsigned mask = ballot32(flag);
  int lane = t & 31, wid = t >> 5;
  int pre = __popc(mask & ((1u << lane) - 1u));
  __shared__ int wtot[32];
  __shared__ int woff[32];
  if (lane == 0) wtot[wid] = __popc(mask);
  __syncthreads();
  if (t == 0) {
    int acc = 0;
    for (int i = 0; i < 32; ++i) { woff[i] = acc; acc += wtot[i]; }
    cnt[e] = acc;
  }
  __syncthreads();
  if (flag) {
    int pos = woff[wid] + pre;
    list[e * T_TOK + pos] = t | (slot << 16);
  }
}

__global__ void prefix16_k(const int* __restrict__ cnt, int* __restrict__ offs) {
  if (threadIdx.x == 0 && blockIdx.x == 0) {
    int a = 0;
    for (int i = 0; i < N_EXP; ++i) { offs[i] = a; a += cnt[i]; }
  }
}

// ---------- 4) GEMM1: h = gather(x) @ w1[e]^T, fused SiLU(gate)*up -> act ----------
// Block tile: BM=64 rows x BN=64 cols of BOTH gate and up halves. 8 waves:
// wave w -> m-tile (w&3), col-pair (w>>2)*2. K staged in LDS 32 at a time as bf16.

__global__ __launch_bounds__(256) void moe_gemm1_k(
    const float* __restrict__ x, const float* __restrict__ w1,
    const int* __restrict__ list, const int* __restrict__ cnt, const int* __restrict__ offs,
    float* __restrict__ act) {
  int e = blockIdx.x;
  int count = cnt[e];
  int m0 = blockIdx.y * 64;
  if (m0 >= count) return;
  int n0 = blockIdx.z * 64;                  // within [0, F_DIM)

  __shared__ alignas(16) unsigned As[64][20];
  __shared__ alignas(16) unsigned Bg[64][20];
  __shared__ alignas(16) unsigned Bu[64][20];
  __shared__ int rows[64];

  int tid = threadIdx.x;
  if (tid < 64) {
    int p = m0 + tid;
    rows[tid] = (p < count) ? (list[e * T_TOK + p] & 0xFFFF) : -1;
  }
  __syncthreads();

  int lane = tid & 31, w = tid >> 5;
  int mi = (w & 3) * 16;                     // local m base
  int nj = (w >> 2) * 2;                     // 16-col tile pair base
  v8f accg0 = {}, accg1 = {}, accu0 = {}, accu1 = {};

  int r = tid >> 2, grp = tid & 3;           // staging role: row r, k-octet grp
  int tok = rows[r];
  const float* xrow = (tok >= 0) ? (x + (size_t)tok * H_DIM) : nullptr;
  const float* w1e  = w1 + (size_t)e * (2 * F_DIM) * H_DIM;
  const float* grow = w1e + (size_t)(n0 + r) * H_DIM;
  const float* urow = w1e + (size_t)(F_DIM + n0 + r) * H_DIM;

  for (int k0 = 0; k0 < H_DIM; k0 += 32) {
    __syncthreads();                         // previous iteration's readers done
    int kk = k0 + grp * 8;
    float4 a0 = {}, a1 = {};
    if (xrow) { a0 = *(const float4*)(xrow + kk); a1 = *(const float4*)(xrow + kk + 4); }
    float4 g0 = *(const float4*)(grow + kk), g1 = *(const float4*)(grow + kk + 4);
    float4 u0 = *(const float4*)(urow + kk), u1 = *(const float4*)(urow + kk + 4);
    int c = grp * 4;
    As[r][c + 0] = bf16pk(a0.x, a0.y); As[r][c + 1] = bf16pk(a0.z, a0.w);
    As[r][c + 2] = bf16pk(a1.x, a1.y); As[r][c + 3] = bf16pk(a1.z, a1.w);
    Bg[r][c + 0] = bf16pk(g0.x, g0.y); Bg[r][c + 1] = bf16pk(g0.z, g0.w);
    Bg[r][c + 2] = bf16pk(g1.x, g1.y); Bg[r][c + 3] = bf16pk(g1.z, g1.w);
    Bu[r][c + 0] = bf16pk(u0.x, u0.y); Bu[r][c + 1] = bf16pk(u0.z, u0.w);
    Bu[r][c + 2] = bf16pk(u1.x, u1.y); Bu[r][c + 3] = bf16pk(u1.z, u1.w);
    if (k0 + 32 < H_DIM) {                   // stream next weight lines (global_prefetch_b8)
      __builtin_prefetch(grow + kk + 32, 0, 1);
      __builtin_prefetch(urow + kk + 32, 0, 1);
    }
    __syncthreads();

    v16bf a = lda_frag(As, mi, lane);
    accg0 = wmma_bf16(a, ldb_frag(Bg, (nj + 0) * 16, lane), accg0);
    accg1 = wmma_bf16(a, ldb_frag(Bg, (nj + 1) * 16, lane), accg1);
    accu0 = wmma_bf16(a, ldb_frag(Bu, (nj + 0) * 16, lane), accu0);
    accu1 = wmma_bf16(a, ldb_frag(Bu, (nj + 1) * 16, lane), accu1);
  }

  // Epilogue: SiLU(gate)*up, scatter to compacted act rows.
  // D layout: lane L holds N = L%16, VGPR rr holds M = rr + 8*(L/16).
  int off_e = offs[e];
  int h = lane >> 4, nlane = lane & 15;
#pragma unroll
  for (int j = 0; j < 2; ++j) {
    v8f g = (j == 0) ? accg0 : accg1;
    v8f u = (j == 0) ? accu0 : accu1;
    int col = n0 + (nj + j) * 16 + nlane;
#pragma unroll
    for (int rr = 0; rr < 8; ++rr) {
      int p = m0 + mi + h * 8 + rr;
      if (p < count) {
        float gv = g[rr];
        float s  = gv / (1.f + __expf(-gv));
        act[(size_t)(off_e + p) * F_DIM + col] = s * u[rr];
      }
    }
  }
}

// ---------- 5) GEMM2: expert_out = act @ w2[e]^T -> rowbuf[token][slot][:] ----------

__global__ __launch_bounds__(256) void moe_gemm2_k(
    const float* __restrict__ act, const float* __restrict__ w2,
    const int* __restrict__ list, const int* __restrict__ cnt, const int* __restrict__ offs,
    float* __restrict__ rowbuf) {
  int e = blockIdx.x;
  int count = cnt[e];
  int m0 = blockIdx.y * 64;
  if (m0 >= count) return;
  int n0 = blockIdx.z * 64;                  // within [0, H_DIM)

  __shared__ alignas(16) unsigned As[64][20];
  __shared__ alignas(16) unsigned Bt[64][20];
  __shared__ int ent[64];

  int tid = threadIdx.x;
  if (tid < 64) {
    int p = m0 + tid;
    ent[tid] = (p < count) ? list[e * T_TOK + p] : -1;
  }
  __syncthreads();

  int lane = tid & 31, w = tid >> 5;
  int mi = (w & 3) * 16;
  int nj = (w >> 2) * 2;
  v8f acc0 = {}, acc1 = {};

  int r = tid >> 2, grp = tid & 3;
  const float* arow = (ent[r] >= 0) ? (act + (size_t)(offs[e] + m0 + r) * F_DIM) : nullptr;
  const float* brow = w2 + ((size_t)e * H_DIM + (n0 + r)) * F_DIM;

  for (int k0 = 0; k0 < F_DIM; k0 += 32) {
    __syncthreads();
    int kk = k0 + grp * 8;
    float4 a0 = {}, a1 = {};
    if (arow) { a0 = *(const float4*)(arow + kk); a1 = *(const float4*)(arow + kk + 4); }
    float4 b0 = *(const float4*)(brow + kk), b1 = *(const float4*)(brow + kk + 4);
    int c = grp * 4;
    As[r][c + 0] = bf16pk(a0.x, a0.y); As[r][c + 1] = bf16pk(a0.z, a0.w);
    As[r][c + 2] = bf16pk(a1.x, a1.y); As[r][c + 3] = bf16pk(a1.z, a1.w);
    Bt[r][c + 0] = bf16pk(b0.x, b0.y); Bt[r][c + 1] = bf16pk(b0.z, b0.w);
    Bt[r][c + 2] = bf16pk(b1.x, b1.y); Bt[r][c + 3] = bf16pk(b1.z, b1.w);
    if (k0 + 32 < F_DIM) __builtin_prefetch(brow + kk + 32, 0, 1);
    __syncthreads();

    v16bf a = lda_frag(As, mi, lane);
    acc0 = wmma_bf16(a, ldb_frag(Bt, (nj + 0) * 16, lane), acc0);
    acc1 = wmma_bf16(a, ldb_frag(Bt, (nj + 1) * 16, lane), acc1);
  }

  int h = lane >> 4, nlane = lane & 15;
#pragma unroll
  for (int j = 0; j < 2; ++j) {
    v8f d = (j == 0) ? acc0 : acc1;
    int col = n0 + (nj + j) * 16 + nlane;
#pragma unroll
    for (int rr = 0; rr < 8; ++rr) {
      int ml = mi + h * 8 + rr;
      int p = m0 + ml;
      if (p < count) {
        int en = ent[ml];
        int t = en & 0xFFFF, slot = en >> 16;
        rowbuf[((size_t)(t * TOPK + slot)) * H_DIM + col] = d[rr];
      }
    }
  }
}

// ---------- 6) combine: out[t,:] = sum_slot w[t,slot] * rowbuf[t,slot,:] ----------

__global__ __launch_bounds__(256) void combine_k(
    const float* __restrict__ rowbuf, const float* __restrict__ topk_w,
    float* __restrict__ out) {
  int idx = blockIdx.x * 256 + threadIdx.x;
  if (idx >= T_TOK * H_DIM) return;
  int t = idx >> 10, col = idx & (H_DIM - 1);
  float s = 0.f;
#pragma unroll
  for (int k = 0; k < TOPK; ++k)
    s += topk_w[t * TOPK + k] * rowbuf[((size_t)(t * TOPK + k)) * H_DIM + col];
  out[idx] = s;
}

// ---------- launch ----------

extern "C" void kernel_launch(void* const* d_in, const int* in_sizes, int n_in,
                              void* d_out, int out_size, void* d_ws, size_t ws_size,
                              hipStream_t stream) {
  const float* x      = (const float*)d_in[0];   // [T,H]
  const float* gate_w = (const float*)d_in[1];   // [E,H]
  const float* w1     = (const float*)d_in[2];   // [E,2F,H]
  const float* w2     = (const float*)d_in[3];   // [E,H,F]
  float* out = (float*)d_out;                    // [T,H]

  char* ws = (char*)d_ws;
  size_t o = 0;
  auto alloc = [&](size_t bytes) -> void* {
    void* p = ws + o;
    o = (o + bytes + 255) & ~(size_t)255;
    return p;
  };
  float* logits  = (float*)alloc((size_t)T_TOK * N_EXP * 4);
  int*   topk_id = (int*)  alloc((size_t)T_TOK * TOPK * 4);
  float* topk_w  = (float*)alloc((size_t)T_TOK * TOPK * 4);
  int*   cnt     = (int*)  alloc(N_EXP * 4);
  int*   offs    = (int*)  alloc(N_EXP * 4);
  int*   list    = (int*)  alloc((size_t)N_EXP * T_TOK * 4);
  float* act     = (float*)alloc((size_t)T_TOK * TOPK * F_DIM * 4);   // 12.6 MB
  float* rowbuf  = (float*)alloc((size_t)T_TOK * TOPK * H_DIM * 4);   // 16.8 MB

  router_logits_k<<<(T_TOK * N_EXP + 255) / 256, 256, 0, stream>>>(x, gate_w, logits);
  router_topk_k<<<(T_TOK + 255) / 256, 256, 0, stream>>>(logits, topk_id, topk_w);
  build_lists_k<<<N_EXP, 1024, 0, stream>>>(topk_id, list, cnt);
  prefix16_k<<<1, 32, 0, stream>>>(cnt, offs);
  moe_gemm1_k<<<dim3(N_EXP, T_TOK / 64, F_DIM / 64), 256, 0, stream>>>(
      x, w1, list, cnt, offs, act);
  moe_gemm2_k<<<dim3(N_EXP, T_TOK / 64, H_DIM / 64), 256, 0, stream>>>(
      act, w2, list, cnt, offs, rowbuf);
  combine_k<<<(T_TOK * H_DIM + 255) / 256, 256, 0, stream>>>(rowbuf, topk_w, out);
}
